// Custom_GCN_59304908423673
// MI455X (gfx1250) — matrix operand order repaired and checked by reading, not compiled, not run on hardware
//
#include <hip/hip_runtime.h>
#include <hip/hip_bf16.h>

typedef __attribute__((ext_vector_type(16))) __bf16 v16bf;
typedef __attribute__((ext_vector_type(8)))  float  v8f;
typedef __attribute__((ext_vector_type(4)))  int    v4i;

#define BM 128
#define BN 256
#define BK 32
#define LDS_STRIDE 40   // elements; 80B rows -> 16B-aligned vector copies, conflict-free frag reads

// pointer-to-int4 in global (AS1) / LDS (AS3) address spaces, per the builtin's signature
typedef __attribute__((address_space(1))) v4i* gp128;
typedef __attribute__((address_space(3))) v4i* lp128;

// ---- CDNA5 async global->LDS copy (ASYNCcnt-tracked), 16 bytes per call ----
__device__ __forceinline__ void async_copy_b128(void* lds, const void* g) {
  __builtin_amdgcn_global_load_async_to_lds_b128((gp128)(void*)g, (lp128)lds, 0, 0);
}

__device__ __forceinline__ void async_wait0() {
#if __has_builtin(__builtin_amdgcn_s_wait_asynccnt)
  __builtin_amdgcn_s_wait_asynccnt(0);
#else
  asm volatile("s_wait_asynccnt 0x0" ::: "memory");
#endif
}

// ---------------- fp32 -> bf16 conversion ----------------
__global__ void f32_to_bf16_kernel(const float* __restrict__ in,
                                   __bf16* __restrict__ out, long long n) {
  long long i = (long long)blockIdx.x * blockDim.x + threadIdx.x;
  long long stride = (long long)gridDim.x * blockDim.x;
  for (; i < n; i += stride) out[i] = (__bf16)in[i];
}

// fp32 [rows][cols] -> bf16 transposed [cols][rows]  (used once for the 512x512 weights)
__global__ void f32_to_bf16_T_kernel(const float* __restrict__ in,
                                     __bf16* __restrict__ out, int rows, int cols) {
  long long i = (long long)blockIdx.x * blockDim.x + threadIdx.x;
  long long n = (long long)rows * cols;
  long long stride = (long long)gridDim.x * blockDim.x;
  for (; i < n; i += stride) {
    int r = (int)(i / cols), c = (int)(i % cols);
    out[(long long)c * rows + r] = (__bf16)in[i];
  }
}

// ---------------- WMMA fragment gathers from LDS ----------------
// A fragment: 16x32 bf16; lanes 0-15: M=lane, K pairs {0..7,16..23}; lanes 16-31: K pairs {8..15,24..31}
__device__ __forceinline__ v16bf load_a_frag(const __bf16* T, int rowBase,
                                             int laneHalf, int lane15) {
  union { v16bf v; unsigned u[8]; } f;
  const int row = rowBase + lane15;
  const int kb = laneHalf << 3;
#pragma unroll
  for (int p = 0; p < 8; ++p) {
    const int k = ((p >> 2) << 4) + kb + ((p & 3) << 1);
    f.u[p] = *(const unsigned*)(T + row * LDS_STRIDE + k);
  }
  return f.v;
}

// B fragment: 32x16 bf16; lane = N column; lanes 0-15: K {0..7,16..23}, lanes 16-31: K {8..15,24..31}.
// LDS tile is already N-major: Bs[n][k].
__device__ __forceinline__ v16bf load_b_frag(const __bf16* T, int colBase,
                                             int laneHalf, int lane15) {
  union { v16bf v; unsigned u[8]; } f;
  const int col = colBase + lane15;
  const int kb = laneHalf << 3;
#pragma unroll
  for (int p = 0; p < 8; ++p)
    f.u[p] = *(const unsigned*)(T + col * LDS_STRIDE + ((p >> 2) << 4) + kb + ((p & 3) << 1));
  return f.v;
}

// ---------------- tiled bf16 WMMA GEMM ----------------
// C = A * BT^T (+bias)(+relu).  A: [M,K] row-major, BT: [N,K] row-major (i.e. B stored N-major).
// Block tile 128x256, 8 waves in 2(M)x4(N); each wave: 64x64 = 16 WMMA tiles.
// TRANS_OUT: store C transposed ([N,M], bf16 only) with fully vectorized b128 stores.
// Batched via blockIdx.z with element strides aBatch/bBatch/cBatch.
template <bool RELU, bool HAS_BIAS, bool TRANS_OUT, typename OutT>
__global__ __launch_bounds__(256) void gemm_bf16_wmma(
    const __bf16* __restrict__ A, int lda, long long aBatch,
    const __bf16* __restrict__ BT, int ldb, long long bBatch,
    const float* __restrict__ bias,
    OutT* __restrict__ C, int ldc, long long cBatch, int K) {
  __shared__ __bf16 As[2][BM * LDS_STRIDE];
  __shared__ __bf16 Bs[2][BN * LDS_STRIDE];

  const int bz = blockIdx.z;
  A  += (long long)bz * aBatch;
  BT += (long long)bz * bBatch;
  C  += (long long)bz * cBatch;

  const int tid      = threadIdx.x;
  const int lane     = tid & 31;       // wave32
  const int wave     = tid >> 5;       // 8 waves
  const int waveM    = wave & 1;       // 2 waves along M (64 rows each)
  const int waveN    = wave >> 1;      // 4 waves along N (64 cols each)
  const int lane15   = lane & 15;
  const int laneHalf = lane >> 4;
  const int blockM   = blockIdx.y * BM;
  const int blockN   = blockIdx.x * BN;

  // per-thread tile-copy coordinates: row t>>1, 8 contiguous K elements (one 16B copy) at K offset 0 or 16
  const int cpR = tid >> 1;            // 0..127
  const int cpC = (tid & 1) << 4;      // 0 or 16
  const __bf16* gA  = A  + (long long)(blockM + cpR) * lda + cpC;
  const __bf16* gB0 = BT + (long long)(blockN + cpR) * ldb + cpC;
  const __bf16* gB1 = BT + (long long)(blockN + 128 + cpR) * ldb + cpC;

  v8f acc[4][4];
#pragma unroll
  for (int tm = 0; tm < 4; ++tm)
#pragma unroll
    for (int tn = 0; tn < 4; ++tn) acc[tm][tn] = (v8f){0.f,0.f,0.f,0.f,0.f,0.f,0.f,0.f};

  // prologue: async-load tile 0 into buffer 0
  async_copy_b128(&As[0][cpR * LDS_STRIDE + cpC], gA);
  async_copy_b128(&Bs[0][cpR * LDS_STRIDE + cpC], gB0);
  async_copy_b128(&Bs[0][(128 + cpR) * LDS_STRIDE + cpC], gB1);
  async_wait0();
  __syncthreads();

  const int nk = K / BK;
  for (int ik = 0; ik < nk; ++ik) {
    const int buf = ik & 1;
    // prefetch next tile into the other buffer (overlaps the WMMAs below)
    if (ik + 1 < nk) {
      const int ko = (ik + 1) * BK;
      async_copy_b128(&As[buf ^ 1][cpR * LDS_STRIDE + cpC], gA + ko);
      async_copy_b128(&Bs[buf ^ 1][cpR * LDS_STRIDE + cpC], gB0 + ko);
      async_copy_b128(&Bs[buf ^ 1][(128 + cpR) * LDS_STRIDE + cpC], gB1 + ko);
    }

    v16bf afrag[4], bfrag[4];
#pragma unroll
    for (int tm = 0; tm < 4; ++tm)
      afrag[tm] = load_a_frag(As[buf], waveM * 64 + tm * 16, laneHalf, lane15);
#pragma unroll
    for (int tn = 0; tn < 4; ++tn)
      bfrag[tn] = load_b_frag(Bs[buf], waveN * 64 + tn * 16, laneHalf, lane15);

#pragma unroll
    for (int tm = 0; tm < 4; ++tm)
#pragma unroll
      for (int tn = 0; tn < 4; ++tn)
        acc[tm][tn] = __builtin_amdgcn_wmma_f32_16x16x32_bf16(
            false, afrag[tm], false, bfrag[tn], (short)0, acc[tm][tn], false, false);

    async_wait0();       // next-buffer tiles landed
    __syncthreads();     // all waves done reading current buffer
  }

  // ---- epilogue. D layout (16x16 f32): VGPR r -> M = r + 8*laneHalf, N = lane15 ----
#pragma unroll
  for (int tm = 0; tm < 4; ++tm) {
#pragma unroll
    for (int tn = 0; tn < 4; ++tn) {
      const int rowBase = blockM + waveM * 64 + tm * 16 + (laneHalf << 3);
      const int col     = blockN + waveN * 64 + tn * 16 + lane15;
      if constexpr (TRANS_OUT) {
        // lane's 8 rows are contiguous in C^T[col][row] -> one 16B store
        union { uint4 v; __bf16 h[8]; } pk;
#pragma unroll
        for (int r = 0; r < 8; ++r) pk.h[r] = (__bf16)acc[tm][tn][r];
        *(uint4*)((__bf16*)C + (long long)col * ldc + rowBase) = pk.v;
      } else {
        float bv = 0.0f;
        if constexpr (HAS_BIAS) bv = bias[col];
#pragma unroll
        for (int r = 0; r < 8; ++r) {
          float v = acc[tm][tn][r] + bv;
          if constexpr (RELU) v = fmaxf(v, 0.0f);
          C[(long long)(rowBase + r) * ldc + col] = (OutT)v;
        }
      }
    }
  }
}

// ---------------- host-side orchestration ----------------
extern "C" void kernel_launch(void* const* d_in, const int* in_sizes, int n_in,
                              void* d_out, int out_size, void* d_ws, size_t ws_size,
                              hipStream_t stream) {
  (void)in_sizes; (void)n_in; (void)out_size; (void)ws_size;
  constexpr int B = 32, N = 512, D = 512;
  constexpr int M = B * N;                       // 16384
  constexpr long long NB  = (long long)M * D;    // 8,388,608
  constexpr long long WSZ = (long long)D * D;    // 262,144

  const float* x   = (const float*)d_in[0];
  const float* adj = (const float*)d_in[1];
  const float* W0  = (const float*)d_in[2];
  const float* b0  = (const float*)d_in[3];
  const float* W1  = (const float*)d_in[4];
  const float* b1  = (const float*)d_in[5];
  const float* W2  = (const float*)d_in[6];
  const float* b2  = (const float*)d_in[7];
  float* out = (float*)d_out;

  // bf16 scratch (~65.5 MB): x, adj, 3x transposed weights, suppT [D][M], h [M][D]
  __bf16* xbf   = (__bf16*)d_ws;
  __bf16* adjbf = xbf + NB;
  __bf16* wtbf  = adjbf + NB;       // W^T, 3 matrices back-to-back
  __bf16* suppT = wtbf + 3 * WSZ;   // [D][M] = [512][16384]
  __bf16* hbf   = suppT + NB;       // [M][D] row-major

  f32_to_bf16_kernel<<<4096, 256, 0, stream>>>(x,   xbf,   NB);
  f32_to_bf16_kernel<<<4096, 256, 0, stream>>>(adj, adjbf, NB);
  f32_to_bf16_T_kernel<<<256, 256, 0, stream>>>(W0, wtbf,           D, D);
  f32_to_bf16_T_kernel<<<256, 256, 0, stream>>>(W1, wtbf + WSZ,     D, D);
  f32_to_bf16_T_kernel<<<256, 256, 0, stream>>>(W2, wtbf + 2 * WSZ, D, D);

  const dim3 gridA(D / BN, M / BM, 1);   // (2, 128, 1): h @ W as one 16384x512x512 GEMM
  const dim3 gridB(D / BN, N / BM, B);   // (2, 4, 32):  batched adj @ support
  const long long S = (long long)N * D;

  // layer 0: suppT = (x @ W0)^T ; h = relu(adj @ supp + b0)
  gemm_bf16_wmma<false, false, true, __bf16><<<gridA, 256, 0, stream>>>(
      xbf, D, 0,  wtbf, D, 0,  nullptr,  suppT, M, 0,  D);
  gemm_bf16_wmma<true, true, false, __bf16><<<gridB, 256, 0, stream>>>(
      adjbf, N, S,  suppT, M, (long long)N,  b0,  hbf, D, S,  N);

  // layer 1
  gemm_bf16_wmma<false, false, true, __bf16><<<gridA, 256, 0, stream>>>(
      hbf, D, 0,  wtbf + WSZ, D, 0,  nullptr,  suppT, M, 0,  D);
  gemm_bf16_wmma<true, true, false, __bf16><<<gridB, 256, 0, stream>>>(
      adjbf, N, S,  suppT, M, (long long)N,  b1,  hbf, D, S,  N);

  // layer 2 (no relu, f32 output)
  gemm_bf16_wmma<false, false, true, __bf16><<<gridA, 256, 0, stream>>>(
      hbf, D, 0,  wtbf + 2 * WSZ, D, 0,  nullptr,  suppT, M, 0,  D);
  gemm_bf16_wmma<false, true, false, float><<<gridB, 256, 0, stream>>>(
      adjbf, N, S,  suppT, M, (long long)N,  b2,  out, D, S,  N);
}
